// YoloV3Loss_52338471469430
// MI455X (gfx1250) — compile-verified, matrix operand order checked by reading, not run on hardware
//
#include <hip/hip_runtime.h>
#include <math.h>

// ---------------------------------------------------------------------------
// YOLOv3 loss, fused single pass per pyramid level + WMMA-based wave reduction.
// d_out: [total, loss_giou, loss_conf, loss_cls] (float32)
// ---------------------------------------------------------------------------

typedef float     v2f  __attribute__((ext_vector_type(2)));
typedef float     v8f  __attribute__((ext_vector_type(8)));
typedef _Float16  v16h __attribute__((ext_vector_type(16)));

#define YOLO_EPS  1e-6f
#define YOLO_NB   150
#define YOLO_NC   80
#define YOLO_IOU_THR 0.5f

__device__ __forceinline__ float softplusf(float x) {
  // jax.nn.softplus(x) == logaddexp(x, 0) == max(x,0) + log1p(exp(-|x|))
  return fmaxf(x, 0.0f) + log1pf(expf(-fabsf(x)));
}

// Full-precision wave32 sum via WMMA.
// Primary: V_WMMA_F32_16X16X4_F32.
//   A layout (16x4 f32): lanes 0-15 -> {V0:K=0, V1:K=1}, lanes 16-31 -> {V0:K=2, V1:K=3}.
//   With A.V0 = x, A.V1 = 0 and B = ones:  D[m][n] = x_m + x_{m+16}.
//   D layout: VGPR r / lanes 0-15 -> M=r, lanes 16-31 -> M=8+r. So:
//     sum8(D) in lanes 0-15  = sum_{m=0..7}  (x_m + x_{m+16})
//     sum8(D) in lanes 16-31 = sum_{m=8..15} (x_m + x_{m+16})
//   One shfl_xor(16) completes the 32-lane sum in every lane.
// Fallback: V_WMMA_F32_16X16X32_F16 with x split into f16 hi+lo (error ~2^-21·x),
//   hi in A element 0 (K=0 / K=8) and lo in element 1; same D reduction.
__device__ __forceinline__ float wave_sum_f32(float x) {
#if defined(__HIP_DEVICE_COMPILE__)
#if __has_builtin(__builtin_amdgcn_wmma_f32_16x16x4_f32)
  v2f a; a[0] = x;    a[1] = 0.0f;
  v2f b; b[0] = 1.0f; b[1] = 1.0f;   // all-ones B regardless of exact K-striping
  v8f c = {};
  v8f d = __builtin_amdgcn_wmma_f32_16x16x4_f32(
      /*neg_a=*/false, a, /*neg_b=*/false, b,
      /*c_mod=*/(short)0, c, /*reuse_a=*/false, /*reuse_b=*/false);
  float s = d[0] + d[1] + d[2] + d[3] + d[4] + d[5] + d[6] + d[7];
  s += __shfl_xor(s, 16, 32);
  return s;
#else
  _Float16 hi = (_Float16)x;
  _Float16 lo = (_Float16)(x - (float)hi);
  v16h a = {};
  a[0] = hi; a[1] = lo;
  v16h b;
  #pragma unroll
  for (int i = 0; i < 16; ++i) b[i] = (_Float16)1.0f;
  v8f c = {};
  v8f d = __builtin_amdgcn_wmma_f32_16x16x32_f16(
      /*neg_a=*/false, a, /*neg_b=*/false, b,
      /*c_mod=*/(short)0, c, /*reuse_a=*/false, /*reuse_b=*/false);
  float s = d[0] + d[1] + d[2] + d[3] + d[4] + d[5] + d[6] + d[7];
  s += __shfl_xor(s, 16, 32);
  return s;
#endif
#else
  return x;  // host stub; never executed
#endif
}

__global__ void __launch_bounds__(256) yolo_zero_kernel(float* out) {
  if (threadIdx.x < 4) out[threadIdx.x] = 0.0f;
}

__global__ void __launch_bounds__(256) yolo_total_kernel(float* out) {
  if (threadIdx.x == 0) out[0] = out[1] + out[2] + out[3];
}

__global__ void __launch_bounds__(256) yolo_layer_kernel(
    const float* __restrict__ p,       // (B, n_anchor, 85)
    const float* __restrict__ p_d,     // (B, n_anchor, 85)
    const float* __restrict__ label,   // (B, n_anchor, 86)
    const float* __restrict__ bboxes,  // (B, 150, 4)
    int n_anchor, float inv_b, float* __restrict__ out)
{
  __shared__ float4 sbb[YOLO_NB];  // candidate boxes for this batch (2.4 KB)
  __shared__ float  red[8 * 3];    // per-wave partials

  const int tid = threadIdx.x;
  const int bat = blockIdx.y;

  if (tid < YOLO_NB) {
    // 16B-aligned: batch stride = 150*16 bytes
    sbb[tid] = reinterpret_cast<const float4*>(bboxes)[bat * YOLO_NB + tid];
  }
  __syncthreads();

  const int  idx   = blockIdx.x * 256 + tid;
  const bool valid = idx < n_anchor;
  const size_t base = (size_t)bat * (size_t)n_anchor + (size_t)(valid ? idx : 0);

  const float* __restrict__ pd = p_d   + base * 85;
  const float* __restrict__ lb = label + base * 86;
  const float* __restrict__ pp = p     + base * 85;

  const float px = pd[0], py = pd[1], pw = pd[2], ph = pd[3];
  const float lx = lb[0], ly = lb[1], lw = lb[2], lh = lb[3];
  const float obj = lb[4], mix = lb[5];
  const float pconf = pp[4];

  // --- predicted box xyxy (raw, for iou_xywh path) ---
  const float p_x1 = px - pw * 0.5f, p_y1 = py - ph * 0.5f;
  const float p_x2 = px + pw * 0.5f, p_y2 = py + ph * 0.5f;

  // --- GIoU (with corner reordering, as in reference) ---
  const float a_x1 = fminf(p_x1, p_x2), a_x2 = fmaxf(p_x1, p_x2);
  const float a_y1 = fminf(p_y1, p_y2), a_y2 = fmaxf(p_y1, p_y2);
  const float l_x1 = lx - lw * 0.5f, l_y1 = ly - lh * 0.5f;
  const float l_x2 = lx + lw * 0.5f, l_y2 = ly + lh * 0.5f;
  const float b_x1 = fminf(l_x1, l_x2), b_x2 = fmaxf(l_x1, l_x2);
  const float b_y1 = fminf(l_y1, l_y2), b_y2 = fmaxf(l_y1, l_y2);

  const float area_a = (a_x2 - a_x1) * (a_y2 - a_y1);
  const float area_b = (b_x2 - b_x1) * (b_y2 - b_y1);
  const float giw = fmaxf(fminf(a_x2, b_x2) - fmaxf(a_x1, b_x1), 0.0f);
  const float gih = fmaxf(fminf(a_y2, b_y2) - fmaxf(a_y1, b_y1), 0.0f);
  const float ginter = giw * gih;
  const float guni   = area_a + area_b - ginter;
  const float giou_i = ginter / fmaxf(guni, YOLO_EPS);
  const float gew = fmaxf(fmaxf(a_x2, b_x2) - fminf(a_x1, b_x1), 0.0f);
  const float geh = fmaxf(fmaxf(a_y2, b_y2) - fminf(a_y1, b_y1), 0.0f);
  const float gea = gew * geh;
  const float giou = giou_i - (gea - guni) / fmaxf(gea, YOLO_EPS);
  float lg = obj * (1.0f - giou) * mix;

  // --- max IoU vs 150 candidate boxes (LDS-staged) ---
  const float area_p = pw * ph;
  float iou_max = 0.0f;
  #pragma unroll 5
  for (int j = 0; j < YOLO_NB; ++j) {
    const float4 q = sbb[j];
    const float q_x1 = q.x - q.z * 0.5f, q_y1 = q.y - q.w * 0.5f;
    const float q_x2 = q.x + q.z * 0.5f, q_y2 = q.y + q.w * 0.5f;
    const float w2 = fmaxf(fminf(p_x2, q_x2) - fmaxf(p_x1, q_x1), 0.0f);
    const float h2 = fmaxf(fminf(p_y2, q_y2) - fmaxf(p_y1, q_y1), 0.0f);
    const float it = w2 * h2;
    const float u2 = area_p + q.z * q.w - it;
    iou_max = fmaxf(iou_max, it / fmaxf(u2, YOLO_EPS));
  }

  // --- focal objectness loss ---
  const float noobj = (1.0f - obj) * ((iou_max < YOLO_IOU_THR) ? 1.0f : 0.0f);
  const float bce_c = softplusf(pconf) - pconf * obj;
  const float sig   = 1.0f / (1.0f + expf(-pconf));
  const float dte   = obj - sig;
  float lc = (obj + noobj) * (bce_c * dte * dte) * mix;

  // --- class BCE (obj is ~2% dense: whole waves skip via execz branch) ---
  float lk = 0.0f;
  if (valid && obj > 0.0f) {
    float acc = 0.0f;
    #pragma unroll 4
    for (int c2 = 0; c2 < YOLO_NC; ++c2) {
      const float x  = pp[5 + c2];
      const float tt = lb[6 + c2];
      acc += softplusf(x) - x * tt;
    }
    lk = obj * mix * acc;
  }
  if (!valid) { lg = 0.0f; lc = 0.0f; }

  // --- reduction: WMMA wave sums (EXEC is all-1s here), then LDS + atomics ---
  const float wlg = wave_sum_f32(lg);
  const float wlc = wave_sum_f32(lc);
  const float wlk = wave_sum_f32(lk);

  const int wid = tid >> 5;
  if ((tid & 31) == 0) {
    red[wid * 3 + 0] = wlg;
    red[wid * 3 + 1] = wlc;
    red[wid * 3 + 2] = wlk;
  }
  __syncthreads();
  if (tid < 3) {
    float s = 0.0f;
    #pragma unroll
    for (int w = 0; w < 8; ++w) s += red[w * 3 + tid];
    atomicAdd(out + 1 + tid, s * inv_b);
  }
}

extern "C" void kernel_launch(void* const* d_in, const int* in_sizes, int n_in,
                              void* d_out, int out_size, void* d_ws, size_t ws_size,
                              hipStream_t stream) {
  // setup_inputs() dict order:
  // 0:p_s 1:p_d_s 2:label_sbbox 3:p_m 4:p_d_m 5:label_mbbox
  // 6:p_l 7:p_d_l 8:label_lbbox 9:sbboxes 10:mbboxes 11:lbboxes
  const float* p_s   = (const float*)d_in[0];
  const float* p_d_s = (const float*)d_in[1];
  const float* lab_s = (const float*)d_in[2];
  const float* p_m   = (const float*)d_in[3];
  const float* p_d_m = (const float*)d_in[4];
  const float* lab_m = (const float*)d_in[5];
  const float* p_l   = (const float*)d_in[6];
  const float* p_d_l = (const float*)d_in[7];
  const float* lab_l = (const float*)d_in[8];
  const float* sbb   = (const float*)d_in[9];
  const float* mbb   = (const float*)d_in[10];
  const float* lbb   = (const float*)d_in[11];
  float* out = (float*)d_out;

  const int   B    = in_sizes[9] / (YOLO_NB * 4);      // 8
  const float invB = 1.0f / (float)B;
  const int   n_s  = in_sizes[0] / (B * 85);           // 76*76*3
  const int   n_m  = in_sizes[3] / (B * 85);           // 38*38*3
  const int   n_l  = in_sizes[6] / (B * 85);           // 19*19*3

  yolo_zero_kernel<<<dim3(1), dim3(256), 0, stream>>>(out);
  yolo_layer_kernel<<<dim3((n_s + 255) / 256, B), dim3(256), 0, stream>>>(
      p_s, p_d_s, lab_s, sbb, n_s, invB, out);
  yolo_layer_kernel<<<dim3((n_m + 255) / 256, B), dim3(256), 0, stream>>>(
      p_m, p_d_m, lab_m, mbb, n_m, invB, out);
  yolo_layer_kernel<<<dim3((n_l + 255) / 256, B), dim3(256), 0, stream>>>(
      p_l, p_d_l, lab_l, lbb, n_l, invB, out);
  yolo_total_kernel<<<dim3(1), dim3(256), 0, stream>>>(out);
}
